// DEC_41180146434796
// MI455X (gfx1250) — compile-verified, matrix operand order checked by reading, not run on hardware
//
#include <hip/hip_runtime.h>
#include <stdint.h>

// ---------------------------------------------------------------------------
// DGCNN forward for MI455X (gfx1250): all GEMM-shaped work through
// v_wmma_f32_16x16x32_bf16 (wave32), irregular work (knn top-k, max-agg) in
// VALU/LDS.  B=32, N=2048, k=5.
// ---------------------------------------------------------------------------

typedef __attribute__((ext_vector_type(16))) __bf16 v16bf;
typedef __attribute__((ext_vector_type(8)))  float  v8f;

#define DEV __device__ __forceinline__

static constexpr int NB = 32;     // batches
static constexpr int NP = 2048;   // points per batch
static constexpr int KNN = 5;     // neighbors (incl. self)

DEV v8f wmma_bf16(v16bf a, v16bf b, v8f c) {
  // D = A(16x32) * B(32x16) + C, f32 accumulate
  return __builtin_amdgcn_wmma_f32_16x16x32_bf16(
      /*neg_a=*/false, a, /*neg_b=*/false, b,
      /*c_mod=*/(short)0, c, /*reuse_a=*/false, /*reuse_b=*/false);
}

// Load one 16x32 bf16 operand tile from a row-major [16][K-span] buffer.
// Works for A (rows = M) and for B when stored transposed as BT[N][K]
// (rows = N columns of B).  Per CDNA5 ISA 16-bit operand layout:
//   lane l (<16): row l,  K in {0..7,16..23};  lane l+16: row l, K in {8..15,24..31}
//   VGPR j holds K pair kb + (j<4 ? 2j : 2(j-4)+16)
DEV v16bf load_tile(const __bf16* base, int stride, int lane) {
  union { v16bf v; uint32_t u[8]; } r;
  int row = lane & 15;
  int kb  = (lane & 16) ? 8 : 0;
  const uint32_t* p = (const uint32_t*)(base + (size_t)row * (size_t)stride);
#pragma unroll
  for (int j = 0; j < 8; ++j) {
    int k = kb + ((j < 4) ? (2 * j) : (2 * (j - 4) + 16));
    r.u[j] = p[k >> 1];
  }
  return r.v;
}

// C/D tile layout: VGPR r, lane l  ->  M = r + (l<16 ? 0 : 8),  N = l & 15.
// Fused epilogue: y = relu(acc + bias) * s + h, store bf16 (row-major dst tile)
DEV void bn_store_bf16(v8f d, float bb, float ss, float hh,
                       __bf16* dst, int stride, int lane) {
  int col = lane & 15;
  int rb  = (lane & 16) ? 8 : 0;
#pragma unroll
  for (int r = 0; r < 8; ++r) {
    float y = d[r] + bb;
    y = (y > 0.f ? y : 0.f) * ss + hh;
    dst[(size_t)(rb + r) * stride + col] = (__bf16)y;
  }
}

DEV void bn_store_f32(v8f d, float bb, float ss, float hh,
                      float* dst, int stride, int lane) {
  int col = lane & 15;
  int rb  = (lane & 16) ? 8 : 0;
#pragma unroll
  for (int r = 0; r < 8; ++r) {
    float y = d[r] + bb;
    y = (y > 0.f ? y : 0.f) * ss + hh;
    dst[(size_t)(rb + r) * stride + col] = y;
  }
}

// ---------------------------------------------------------------------------
// K0: weight prep — f32 [K][N] -> bf16 transposed [N][Kpad] (zero-padded K)
// ---------------------------------------------------------------------------
__global__ void wtrans_kernel(const float* __restrict__ src, __bf16* __restrict__ dst,
                              int K, int N, int Kpad) {
  int idx = blockIdx.x * blockDim.x + threadIdx.x;
  int total = N * Kpad;
  if (idx >= total) return;
  int o = idx / Kpad;
  int k = idx - o * Kpad;
  dst[idx] = (k < K) ? (__bf16)src[(size_t)k * N + o] : (__bf16)0.0f;
}

// ---------------------------------------------------------------------------
// K1: kNN on raw positions (F=3).  Batch positions staged in LDS (24 KB).
// ---------------------------------------------------------------------------
__global__ __launch_bounds__(256) void knn1_kernel(const float* __restrict__ pos,
                                                   int* __restrict__ idx1) {
  __shared__ float sx[NP * 3];
  int b = blockIdx.x >> 3;
  int chunk = blockIdx.x & 7;
  const float* pb = pos + (size_t)b * NP * 3;
  for (int t = threadIdx.x; t < NP * 3; t += 256) sx[t] = pb[t];
  __syncthreads();

  int i = chunk * 256 + threadIdx.x;
  float x0 = sx[i * 3], x1 = sx[i * 3 + 1], x2 = sx[i * 3 + 2];
  float bd[KNN]; int bi[KNN];
#pragma unroll
  for (int q = 0; q < KNN; ++q) { bd[q] = 3.4e38f; bi[q] = 0; }
  for (int j = 0; j < NP; ++j) {
    float d0 = sx[j * 3] - x0, d1 = sx[j * 3 + 1] - x1, d2 = sx[j * 3 + 2] - x2;
    float d = d0 * d0 + d1 * d1 + d2 * d2;
    if (d < bd[KNN - 1]) {
      int p = KNN - 1;
      while (p > 0 && d < bd[p - 1]) { bd[p] = bd[p - 1]; bi[p] = bi[p - 1]; --p; }
      bd[p] = d; bi[p] = j;
    }
  }
  int* out = idx1 + ((size_t)b * NP + i) * KNN;
#pragma unroll
  for (int q = 0; q < KNN; ++q) out[q] = bi[q];
}

// ---------------------------------------------------------------------------
// K2: EdgeConv1 — 16 points / wave -> 80 edges = 5 M-tiles; MLP 6->64->64->64
// (K padded 6->32), fused bias/relu/bn, max over k=5, write bf16 to feat[0:64]
// ---------------------------------------------------------------------------
__global__ __launch_bounds__(32) void ec1_kernel(
    const float* __restrict__ pos, const int* __restrict__ idx1,
    __bf16* __restrict__ feat,
    const __bf16* __restrict__ w1aT, const __bf16* __restrict__ w1bT,
    const __bf16* __restrict__ w1cT,
    const float* b1a, const float* s1a, const float* h1a,
    const float* b1b, const float* s1b, const float* h1b,
    const float* b1c, const float* s1c, const float* h1c) {
  __shared__ __bf16 eA[80 * 32];
  __shared__ __bf16 hA[80 * 64];
  __shared__ __bf16 hB[80 * 64];
  __shared__ float  fO[80 * 64];

  int lane = threadIdx.x;
  int b = blockIdx.x >> 7;           // 128 groups of 16 points per batch
  int base = (blockIdx.x & 127) * 16;

  // build edge features [xi, xj - xi], pad K 6 -> 32
  for (int e = lane; e < 80; e += 32) {
    int p = base + e / 5;
    int j = idx1[((size_t)b * NP + p) * KNN + (e % 5)];
    const float* xi = pos + ((size_t)b * NP + p) * 3;
    const float* xj = pos + ((size_t)b * NP + j) * 3;
    __bf16* row = eA + e * 32;
    row[0] = (__bf16)xi[0]; row[1] = (__bf16)xi[1]; row[2] = (__bf16)xi[2];
    row[3] = (__bf16)(xj[0] - xi[0]);
    row[4] = (__bf16)(xj[1] - xi[1]);
    row[5] = (__bf16)(xj[2] - xi[2]);
    for (int k = 6; k < 32; ++k) row[k] = (__bf16)0.0f;
  }
  __syncthreads();

  int col = lane & 15;
  // layer 1: 6(->32) -> 64
  for (int m = 0; m < 5; ++m) {
    v16bf a = load_tile(eA + m * 16 * 32, 32, lane);
    for (int n = 0; n < 4; ++n) {
      v16bf bm = load_tile(w1aT + n * 16 * 32, 32, lane);
      v8f c = {};
      c = wmma_bf16(a, bm, c);
      int cg = n * 16 + col;
      bn_store_bf16(c, b1a[cg], s1a[cg], h1a[cg], hA + m * 16 * 64 + n * 16, 64, lane);
    }
  }
  __syncthreads();
  // layer 2: 64 -> 64 (2 k-steps)
  for (int m = 0; m < 5; ++m) {
    v16bf a0 = load_tile(hA + m * 16 * 64 + 0, 64, lane);
    v16bf a1 = load_tile(hA + m * 16 * 64 + 32, 64, lane);
    for (int n = 0; n < 4; ++n) {
      v8f c = {};
      c = wmma_bf16(a0, load_tile(w1bT + n * 16 * 64 + 0, 64, lane), c);
      c = wmma_bf16(a1, load_tile(w1bT + n * 16 * 64 + 32, 64, lane), c);
      int cg = n * 16 + col;
      bn_store_bf16(c, b1b[cg], s1b[cg], h1b[cg], hB + m * 16 * 64 + n * 16, 64, lane);
    }
  }
  __syncthreads();
  // layer 3: 64 -> 64 -> f32 out
  for (int m = 0; m < 5; ++m) {
    v16bf a0 = load_tile(hB + m * 16 * 64 + 0, 64, lane);
    v16bf a1 = load_tile(hB + m * 16 * 64 + 32, 64, lane);
    for (int n = 0; n < 4; ++n) {
      v8f c = {};
      c = wmma_bf16(a0, load_tile(w1cT + n * 16 * 64 + 0, 64, lane), c);
      c = wmma_bf16(a1, load_tile(w1cT + n * 16 * 64 + 32, 64, lane), c);
      int cg = n * 16 + col;
      bn_store_f32(c, b1c[cg], s1c[cg], h1c[cg], fO + m * 16 * 64 + n * 16, 64, lane);
    }
  }
  __syncthreads();
  // max over k=5 edges per point -> feat channels [0,64)
  for (int t = lane; t < 16 * 64; t += 32) {
    int p = t >> 6, ch = t & 63;
    float m = fO[(p * 5 + 0) * 64 + ch];
#pragma unroll
    for (int e = 1; e < 5; ++e) m = fmaxf(m, fO[(p * 5 + e) * 64 + ch]);
    feat[((size_t)b * NP + base + p) * 192 + ch] = (__bf16)m;
  }
}

// ---------------------------------------------------------------------------
// K3: squared norms of x1 (bf16 channels 0..63)
// ---------------------------------------------------------------------------
__global__ void norms_kernel(const __bf16* __restrict__ feat, float* __restrict__ norms) {
  int idx = blockIdx.x * blockDim.x + threadIdx.x;
  if (idx >= NB * NP) return;
  const __bf16* row = feat + (size_t)idx * 192;
  float s = 0.f;
  for (int c = 0; c < 64; ++c) { float v = (float)row[c]; s += v * v; }
  norms[idx] = s;
}

// ---------------------------------------------------------------------------
// K4: kNN on x1 via WMMA GEMM  S = Xi * Xj^T  (K=64, 2 k-steps),
//     d = |xi|^2 + |xj|^2 - 2 S ; per-row top-5 on lanes 0..15
// ---------------------------------------------------------------------------
__global__ __launch_bounds__(32) void knn2_kernel(const __bf16* __restrict__ feat,
                                                  const float* __restrict__ norms,
                                                  int* __restrict__ idx2) {
  __shared__ float sd[16 * 16];
  __shared__ float snj[16];
  int lane = threadIdx.x;
  int b = blockIdx.x >> 7;
  int i0 = (blockIdx.x & 127) * 16;
  const __bf16* fb = feat + (size_t)b * NP * 192;

  // A tiles (Xi rows) held in registers across the whole j loop
  v16bf a0 = load_tile(fb + (size_t)i0 * 192 + 0, 192, lane);
  v16bf a1 = load_tile(fb + (size_t)i0 * 192 + 32, 192, lane);

  float bd[KNN]; int bi[KNN];
#pragma unroll
  for (int q = 0; q < KNN; ++q) { bd[q] = 3.4e38f; bi[q] = 0; }
  float ni = (lane < 16) ? norms[(size_t)b * NP + i0 + lane] : 0.f;

  int col = lane & 15;
  int rb  = (lane & 16) ? 8 : 0;
  for (int jt = 0; jt < 128; ++jt) {
    int j0 = jt * 16;
    v16bf bm0 = load_tile(fb + (size_t)j0 * 192 + 0, 192, lane);
    v16bf bm1 = load_tile(fb + (size_t)j0 * 192 + 32, 192, lane);
    v8f c = {};
    c = wmma_bf16(a0, bm0, c);
    c = wmma_bf16(a1, bm1, c);
    if (lane < 16) snj[lane] = norms[(size_t)b * NP + j0 + lane];
#pragma unroll
    for (int r = 0; r < 8; ++r) sd[(rb + r) * 16 + col] = c[r];
    __syncthreads();
    if (lane < 16) {
      for (int jl = 0; jl < 16; ++jl) {
        float d = ni + snj[jl] - 2.0f * sd[lane * 16 + jl];
        if (d < bd[KNN - 1]) {
          int p = KNN - 1;
          while (p > 0 && d < bd[p - 1]) { bd[p] = bd[p - 1]; bi[p] = bi[p - 1]; --p; }
          bd[p] = d; bi[p] = j0 + jl;
        }
      }
    }
    __syncthreads();
  }
  if (lane < 16) {
    int* out = idx2 + ((size_t)b * NP + i0 + lane) * KNN;
#pragma unroll
    for (int q = 0; q < KNN; ++q) out[q] = bi[q];
  }
}

// ---------------------------------------------------------------------------
// K5: EdgeConv2 — edge feat 128 = [x1_i, x1_j - x1_i], MLP 128->128,
//     max over k, write bf16 into feat channels [64,192)
// ---------------------------------------------------------------------------
__global__ __launch_bounds__(32) void ec2_kernel(
    __bf16* __restrict__ feat, const int* __restrict__ idx2,
    const __bf16* __restrict__ w2T,
    const float* b2, const float* s2, const float* h2) {
  __shared__ __bf16 eA[80 * 128];   // 20 KB
  __shared__ float  fO[80 * 128];   // 40 KB
  int lane = threadIdx.x;
  int b = blockIdx.x >> 7;
  int base = (blockIdx.x & 127) * 16;

  for (int e = lane; e < 80; e += 32) {
    int p = base + e / 5;
    int j = idx2[((size_t)b * NP + p) * KNN + (e % 5)];
    const __bf16* xi = feat + ((size_t)b * NP + p) * 192;
    const __bf16* xj = feat + ((size_t)b * NP + j) * 192;
    for (int ch = 0; ch < 64; ++ch) {
      float vi = (float)xi[ch];
      eA[e * 128 + ch] = (__bf16)vi;
      eA[e * 128 + 64 + ch] = (__bf16)((float)xj[ch] - vi);
    }
  }
  __syncthreads();

  int col = lane & 15;
  for (int m = 0; m < 5; ++m) {
    v16bf a[4];
#pragma unroll
    for (int ks = 0; ks < 4; ++ks)
      a[ks] = load_tile(eA + m * 16 * 128 + ks * 32, 128, lane);
    for (int n = 0; n < 8; ++n) {
      v8f c = {};
#pragma unroll
      for (int ks = 0; ks < 4; ++ks)
        c = wmma_bf16(a[ks], load_tile(w2T + n * 16 * 128 + ks * 32, 128, lane), c);
      int cg = n * 16 + col;
      bn_store_f32(c, b2[cg], s2[cg], h2[cg], fO + m * 16 * 128 + n * 16, 128, lane);
    }
  }
  __syncthreads();
  for (int t = lane; t < 16 * 128; t += 32) {
    int p = t >> 7, ch = t & 127;
    float m = fO[(p * 5 + 0) * 128 + ch];
#pragma unroll
    for (int e = 1; e < 5; ++e) m = fmaxf(m, fO[(p * 5 + e) * 128 + ch]);
    feat[((size_t)b * NP + base + p) * 192 + 64 + ch] = (__bf16)m;
  }
}

// ---------------------------------------------------------------------------
// K6: final 192->1024 linear + relu/bn with FUSED global max pool over N.
//     One wave per (batch, 16-col tile), running column max across 128 M-tiles.
// ---------------------------------------------------------------------------
__global__ __launch_bounds__(32) void flk_kernel(
    const __bf16* __restrict__ feat, const __bf16* __restrict__ wlT,
    const float* bl, const float* sl, const float* hl,
    float* __restrict__ pooled, __bf16* __restrict__ pooledB) {
  __shared__ float red[32];
  int lane = threadIdx.x;
  int b = blockIdx.x >> 6;
  int c0 = (blockIdx.x & 63) * 16;
  int col = lane & 15;
  int cg = c0 + col;
  float bb = bl[cg], ss = sl[cg], hh = hl[cg];

  float rmax = -3.402823466e38f;
  for (int mt = 0; mt < 128; ++mt) {
    const __bf16* arow = feat + ((size_t)b * NP + mt * 16) * 192;
    v8f c = {};
#pragma unroll
    for (int ks = 0; ks < 6; ++ks) {
      v16bf a  = load_tile(arow + ks * 32, 192, lane);
      v16bf bm = load_tile(wlT + (size_t)c0 * 192 + ks * 32, 192, lane);
      c = wmma_bf16(a, bm, c);
    }
#pragma unroll
    for (int r = 0; r < 8; ++r) {
      float y = c[r] + bb;
      y = (y > 0.f ? y : 0.f) * ss + hh;
      rmax = fmaxf(rmax, y);
    }
  }
  red[lane] = rmax;
  __syncthreads();
  if (lane < 16) {
    float m = fmaxf(red[lane], red[lane + 16]);
    pooled[(size_t)b * 1024 + cg]  = m;
    pooledB[(size_t)b * 1024 + cg] = (__bf16)m;
  }
}

// ---------------------------------------------------------------------------
// K7/K8: head layers (M=32 rows).  One wave per 16x16 output tile.
// ---------------------------------------------------------------------------
__global__ __launch_bounds__(32) void head_kernel(
    const __bf16* __restrict__ A, const __bf16* __restrict__ BT,
    const float* bias, const float* sc, const float* sh,
    float* __restrict__ outF, __bf16* __restrict__ outB, int K, int Nout) {
  int lane = threadIdx.x;
  int nt = blockIdx.x, mt = blockIdx.y;
  v8f c = {};
  int ksteps = K >> 5;
  for (int ks = 0; ks < ksteps; ++ks) {
    v16bf a  = load_tile(A + (size_t)mt * 16 * K + ks * 32, K, lane);
    v16bf bm = load_tile(BT + (size_t)nt * 16 * K + ks * 32, K, lane);
    c = wmma_bf16(a, bm, c);
  }
  int col = lane & 15;
  int rb  = (lane & 16) ? 8 : 0;
  int cg = nt * 16 + col;
  float bb = bias[cg], ss = sc[cg], hh = sh[cg];
#pragma unroll
  for (int r = 0; r < 8; ++r) {
    float y = c[r] + bb;
    y = (y > 0.f ? y : 0.f) * ss + hh;
    int row = mt * 16 + rb + r;
    outF[(size_t)row * Nout + cg] = y;
    if (outB) outB[(size_t)row * Nout + cg] = (__bf16)y;
  }
}

// K9: 256 -> 2 output head (tiny, scalar)
__global__ void out_kernel(const float* __restrict__ h2f, const float* __restrict__ wout,
                           const float* __restrict__ bout, float* __restrict__ out) {
  int t = blockIdx.x * blockDim.x + threadIdx.x;
  if (t >= NB * 2) return;
  int b = t >> 1, c = t & 1;
  float acc = bout[c];
  for (int k = 0; k < 256; ++k) acc += h2f[b * 256 + k] * wout[k * 2 + c];
  out[t] = acc;
}

// ---------------------------------------------------------------------------
// Workspace layout (all sizes 256B-aligned), ~30 MB total
// ---------------------------------------------------------------------------
static constexpr size_t FEAT_SZ  = (size_t)NB * NP * 192 * 2;   // bf16 [x1|x2]
static constexpr size_t IDX_SZ   = (size_t)NB * NP * KNN * 4;
static constexpr size_t NORM_SZ  = (size_t)NB * NP * 4;
static constexpr size_t POOL_SZ  = (size_t)NB * 1024 * 4;
static constexpr size_t POOLB_SZ = (size_t)NB * 1024 * 2;
static constexpr size_t H1_SZ    = (size_t)NB * 512 * 4;
static constexpr size_t H1B_SZ   = (size_t)NB * 512 * 2;
static constexpr size_t H2_SZ    = (size_t)NB * 256 * 4;

extern "C" void kernel_launch(void* const* d_in, const int* in_sizes, int n_in,
                              void* d_out, int out_size, void* d_ws, size_t ws_size,
                              hipStream_t stream) {
  const float* pos = (const float*)d_in[0];
  const float *w1a = (const float*)d_in[1],  *b1a = (const float*)d_in[2],
              *s1a = (const float*)d_in[3],  *h1a = (const float*)d_in[4];
  const float *w1b = (const float*)d_in[5],  *b1b = (const float*)d_in[6],
              *s1b = (const float*)d_in[7],  *h1b = (const float*)d_in[8];
  const float *w1c = (const float*)d_in[9],  *b1c = (const float*)d_in[10],
              *s1c = (const float*)d_in[11], *h1c = (const float*)d_in[12];
  const float *w2  = (const float*)d_in[13], *b2  = (const float*)d_in[14],
              *s2  = (const float*)d_in[15], *h2  = (const float*)d_in[16];
  const float *wl  = (const float*)d_in[17], *bl  = (const float*)d_in[18],
              *sl  = (const float*)d_in[19], *hl  = (const float*)d_in[20];
  const float *wm1 = (const float*)d_in[21], *bm1 = (const float*)d_in[22],
              *sm1 = (const float*)d_in[23], *hm1 = (const float*)d_in[24];
  const float *wm2 = (const float*)d_in[25], *bm2 = (const float*)d_in[26],
              *sm2 = (const float*)d_in[27], *hm2 = (const float*)d_in[28];
  const float *wout = (const float*)d_in[29], *bout = (const float*)d_in[30];
  float* out = (float*)d_out;

  char* ws = (char*)d_ws;
  size_t off = 0;
  __bf16* feat    = (__bf16*)(ws + off); off += FEAT_SZ;
  int*    idx1    = (int*)   (ws + off); off += IDX_SZ;
  int*    idx2    = (int*)   (ws + off); off += IDX_SZ;
  float*  norms   = (float*) (ws + off); off += NORM_SZ;
  float*  pooled  = (float*) (ws + off); off += POOL_SZ;
  __bf16* pooledB = (__bf16*)(ws + off); off += POOLB_SZ;
  float*  h1f     = (float*) (ws + off); off += H1_SZ;
  __bf16* h1b16   = (__bf16*)(ws + off); off += H1B_SZ;
  float*  h2f     = (float*) (ws + off); off += H2_SZ;
  __bf16* w1aT    = (__bf16*)(ws + off); off += (size_t)64 * 32 * 2;
  __bf16* w1bT    = (__bf16*)(ws + off); off += (size_t)64 * 64 * 2;
  __bf16* w1cT    = (__bf16*)(ws + off); off += (size_t)64 * 64 * 2;
  __bf16* w2T     = (__bf16*)(ws + off); off += (size_t)128 * 128 * 2;
  __bf16* wlT     = (__bf16*)(ws + off); off += (size_t)1024 * 192 * 2;
  __bf16* wm1T    = (__bf16*)(ws + off); off += (size_t)512 * 1024 * 2;
  __bf16* wm2T    = (__bf16*)(ws + off); off += (size_t)256 * 512 * 2;

  auto wprep = [&](const float* s, __bf16* d, int K, int N, int Kpad) {
    int total = N * Kpad;
    wtrans_kernel<<<(total + 255) / 256, 256, 0, stream>>>(s, d, K, N, Kpad);
  };
  wprep(w1a, w1aT, 6, 64, 32);
  wprep(w1b, w1bT, 64, 64, 64);
  wprep(w1c, w1cT, 64, 64, 64);
  wprep(w2,  w2T, 128, 128, 128);
  wprep(wl,  wlT, 192, 1024, 192);
  wprep(wm1, wm1T, 1024, 512, 1024);
  wprep(wm2, wm2T, 512, 256, 512);

  // kNN on pos
  knn1_kernel<<<NB * (NP / 256), 256, 0, stream>>>(pos, idx1);
  // EdgeConv1 -> feat[:,:,0:64]
  ec1_kernel<<<NB * (NP / 16), 32, 0, stream>>>(pos, idx1, feat,
      w1aT, w1bT, w1cT, b1a, s1a, h1a, b1b, s1b, h1b, b1c, s1c, h1c);
  // norms of x1
  norms_kernel<<<(NB * NP + 255) / 256, 256, 0, stream>>>(feat, norms);
  // kNN on x1 (WMMA GEMM distances)
  knn2_kernel<<<NB * (NP / 16), 32, 0, stream>>>(feat, norms, idx2);
  // EdgeConv2 -> feat[:,:,64:192]
  ec2_kernel<<<NB * (NP / 16), 32, 0, stream>>>(feat, idx2, w2T, b2, s2, h2);
  // 192->1024 + fused global max pool
  flk_kernel<<<NB * 64, 32, 0, stream>>>(feat, wlT, bl, sl, hl, pooled, pooledB);
  // head: 1024->512, 512->256
  head_kernel<<<dim3(512 / 16, NB / 16), 32, 0, stream>>>(
      pooledB, wm1T, bm1, sm1, hm1, h1f, h1b16, 1024, 512);
  head_kernel<<<dim3(256 / 16, NB / 16), 32, 0, stream>>>(
      h1b16, wm2T, bm2, sm2, hm2, h2f, nullptr, 512, 256);
  // 256 -> 2
  out_kernel<<<1, 64, 0, stream>>>(h2f, wout, bout, out);
  (void)in_sizes; (void)n_in; (void)out_size; (void)ws_size;
}
